// LLM_6571299963498
// MI455X (gfx1250) — matrix operand order repaired and checked by reading, not compiled, not run on hardware
//
#include <hip/hip_runtime.h>
#include <math.h>

typedef __attribute__((ext_vector_type(16))) _Float16 v16h;
typedef __attribute__((ext_vector_type(8)))  _Float16 v8h;
typedef __attribute__((ext_vector_type(8)))  float    v8f;
typedef __attribute__((ext_vector_type(4)))  float    v4f;

constexpr int Bn=32, Tn=384, CFn=128, Dn=512, Hn=16, HDn=32, Ln=16, HIDn=512;
constexpr int Mrows = Bn*Tn;   // 12288

// ---------------------------------------------------------------- async LDS copy
// CDNA5 GLOBAL_LOAD_ASYNC_TO_LDS_B128: per-lane global->LDS DMA, tracked by
// ASYNCcnt (no VGPR round trip). Signature (from clang diagnostic):
//   (v4i addrspace(1)*, v4i addrspace(3)*, int imm_offset, int cpol)
#if defined(__has_builtin)
#if __has_builtin(__builtin_amdgcn_global_load_async_to_lds_b128)
#define HAVE_ASYNC_LDS 1
#endif
#endif
#ifndef HAVE_ASYNC_LDS
#define HAVE_ASYNC_LDS 0
#endif

#if HAVE_ASYNC_LDS
typedef int v4i_vs __attribute__((vector_size(4 * sizeof(int))));
typedef __attribute__((address_space(1))) v4i_vs g_v4i;
typedef __attribute__((address_space(3))) v4i_vs l_v4i;
#endif

static __device__ __forceinline__ void copy16_to_lds(void* lds_dst, const void* gsrc){
#if HAVE_ASYNC_LDS
  __builtin_amdgcn_global_load_async_to_lds_b128(
      (g_v4i*)gsrc, (l_v4i*)lds_dst, 0, 0);
#else
  *(v8h*)lds_dst = *(const v8h*)gsrc;
#endif
}

static __device__ __forceinline__ void async_lds_wait(){
#if HAVE_ASYNC_LDS
#if defined(__has_builtin) && __has_builtin(__builtin_amdgcn_s_wait_asynccnt)
  __builtin_amdgcn_s_wait_asynccnt(0);
#else
  asm volatile("s_wait_asynccnt 0" ::: "memory");
#endif
#endif
}

// ---------------------------------------------------------------- WMMA helpers
static __device__ __forceinline__ v8f wmma16(v16h a, v16h b, v8f c){
  // D = A(16x32 f16) x B(32x16 f16) + C(16x16 f32)
  return __builtin_amdgcn_wmma_f32_16x16x32_f16(false, a, false, b, (short)0, c,
                                                false, false);
}

// A fragment (16x32) from f16 LDS tile, row-major, stride lda halfs (mult of 8).
// CDNA5 layout: lane l<16 -> M=l, K in [0,8)+[16,24); lane l+16 -> K in [8,16)+[24,32)
static __device__ __forceinline__ v16h load_a_frag(const _Float16* base, int lda){
  const int lane = threadIdx.x & 31;
  const int hi = lane >> 4, l = lane & 15;
  const _Float16* p = base + l*lda + hi*8;
  const v8h lo = *(const v8h*)p;          // K = hi*8 + 0..7
  const v8h hs = *(const v8h*)(p + 16);   // K = 16 + hi*8 + 0..7
  v16h r;
#pragma unroll
  for (int i=0;i<8;i++){ r[i]=lo[i]; r[i+8]=hs[i]; }
  return r;
}

// A fragment from f32 LDS (convert to f16 on the fly), stride lda floats (mult of 8)
static __device__ __forceinline__ v16h load_a_frag_f32(const float* base, int lda){
  const int lane = threadIdx.x & 31;
  const int hi = lane >> 4, l = lane & 15;
  const float* p = base + l*lda + hi*8;
  const v4f a0 = *(const v4f*)p,      a1 = *(const v4f*)(p+4);
  const v4f b0 = *(const v4f*)(p+16), b1 = *(const v4f*)(p+20);
  v16h r;
#pragma unroll
  for (int i=0;i<4;i++){
    r[i]    = (_Float16)a0[i];  r[i+4]  = (_Float16)a1[i];
    r[i+8]  = (_Float16)b0[i];  r[i+12] = (_Float16)b1[i];
  }
  return r;
}

// B fragment (32x16) from N-major f16 LDS (Bt[n][k]), stride ldb halfs (mult of 8).
// lane -> column N=lane&15; elements i<8 -> K = (lane>>4)*8 + i, i>=8 -> K = 16 + (lane>>4)*8 + (i-8)
static __device__ __forceinline__ v16h load_b_frag(const _Float16* base, int ldb){
  const int lane = threadIdx.x & 31;
  const int hi = lane >> 4, l = lane & 15;
  const _Float16* p = base + l*ldb + hi*8;
  const v8h lo = *(const v8h*)p;
  const v8h hs = *(const v8h*)(p + 16);
  v16h r;
#pragma unroll
  for (int i=0;i<8;i++){ r[i]=lo[i]; r[i+8]=hs[i]; }
  return r;
}

// ---------------------------------------------------------------- GEMM (WMMA)
// C[M,N] = act(A[M,K] @ B[K,N] + bias) (+ Res).  A,B,Res f32 row-major.
// Block tile 128x128, 8 waves, wave tile 32x64. Double-buffered LDS: one
// barrier per K-step, global loads for tile k+1 overlap WMMA of tile k.
template<typename OutT, bool RELU, bool RES>
__global__ __launch_bounds__(256) void gemm_wmma(
    const float* __restrict__ A, const float* __restrict__ B,
    const float* __restrict__ bias, const float* __restrict__ Res,
    OutT* __restrict__ C, int M, int N, int K)
{
  constexpr int LDA=40, LDB=40;
  __shared__ _Float16 As[2][128*LDA];   // [m][k]
  __shared__ _Float16 Bt[2][128*LDB];   // [n][k] (transposed)
  const int n0 = blockIdx.x*128, m0 = blockIdx.y*128;
  const int t = threadIdx.x, lane = t&31, w = t>>5;
  const int wm = (w&3)*32, wn = (w>>2)*64;
  const int acol4=(t&7)*4, arow=t>>3;       // A staging pattern
  const int bn4=(t&31)*4,  bkr=t>>5;        // B staging pattern
  v8f acc[2][4] = {};
  float4 aReg[4], bReg[4];

  auto gload = [&](int k0){
#pragma unroll
    for (int r=0;r<4;r++)
      aReg[r] = *(const float4*)(A + (size_t)(m0+arow+r*32)*K + k0 + acol4);
#pragma unroll
    for (int r=0;r<4;r++)
      bReg[r] = *(const float4*)(B + (size_t)(k0+bkr+r*8)*N + n0 + bn4);
  };
  auto lstore = [&](int buf){
#pragma unroll
    for (int r=0;r<4;r++){
      _Float16* d = &As[buf][(arow+r*32)*LDA + acol4];
      d[0]=(_Float16)aReg[r].x; d[1]=(_Float16)aReg[r].y;
      d[2]=(_Float16)aReg[r].z; d[3]=(_Float16)aReg[r].w;
    }
#pragma unroll
    for (int r=0;r<4;r++){
      const int kk = bkr + r*8;
      Bt[buf][(bn4+0)*LDB+kk]=(_Float16)bReg[r].x;
      Bt[buf][(bn4+1)*LDB+kk]=(_Float16)bReg[r].y;
      Bt[buf][(bn4+2)*LDB+kk]=(_Float16)bReg[r].z;
      Bt[buf][(bn4+3)*LDB+kk]=(_Float16)bReg[r].w;
    }
  };

  gload(0); lstore(0);
  const int KT = K >> 5;
  for (int kt=0; kt<KT; kt++){
    if (kt+1 < KT) gload((kt+1)*32);
    __syncthreads();
    const int cur = kt & 1;
    v16h af[2];
#pragma unroll
    for (int i=0;i<2;i++) af[i] = load_a_frag(&As[cur][(wm+16*i)*LDA], LDA);
#pragma unroll
    for (int j=0;j<4;j++){
      const v16h bf = load_b_frag(&Bt[cur][(wn+16*j)*LDB], LDB);
#pragma unroll
      for (int i=0;i<2;i++) acc[i][j] = wmma16(af[i], bf, acc[i][j]);
    }
    if (kt+1 < KT) lstore((kt+1)&1);
  }

  const int hi=lane>>4, l=lane&15;
#pragma unroll
  for (int i=0;i<2;i++)
#pragma unroll
    for (int j=0;j<4;j++){
      const int colg = n0 + wn + 16*j + l;
      const float bv = bias ? bias[colg] : 0.f;
#pragma unroll
      for (int r=0;r<8;r++){
        const int rowg = m0 + wm + 16*i + hi*8 + r;
        float v = acc[i][j][r] + bv;
        if (RES)  v += Res[(size_t)rowg*N + colg];
        if (RELU) v = v>0.f ? v : 0.f;
        C[(size_t)rowg*N + colg] = (OutT)v;
      }
    }
}

// ---------------------------------------------------------------- LayerNorm
__global__ __launch_bounds__(256) void ln_kernel(
    const float* __restrict__ x, const float* __restrict__ g,
    const float* __restrict__ b, float* __restrict__ out)
{
  const int w = threadIdx.x>>5, lane = threadIdx.x&31;
  const int row = blockIdx.x*8 + w;
  const float* xr = x + (size_t)row*Dn;
  float v[16]; float s=0.f;
#pragma unroll
  for (int i=0;i<16;i++){ v[i]=xr[lane + i*32]; s+=v[i]; }
#pragma unroll
  for (int o=16;o;o>>=1) s += __shfl_xor(s,o,32);
  const float mean = s*(1.f/Dn);
  float var=0.f;
#pragma unroll
  for (int i=0;i<16;i++){ const float d=v[i]-mean; var+=d*d; }
#pragma unroll
  for (int o=16;o;o>>=1) var += __shfl_xor(var,o,32);
  const float inv = rsqrtf(var*(1.f/Dn) + 1e-6f);
  float* orow = out + (size_t)row*Dn;
#pragma unroll
  for (int i=0;i<16;i++){
    const int c = lane + i*32;
    orow[c] = (v[i]-mean)*inv*g[c] + b[c];
  }
}

// ---------------------------------------------------------------- misc elementwise
__global__ void addpe_kernel(float* __restrict__ x, const float* __restrict__ pe,
                             const float* __restrict__ lpe)
{
  const size_t i = (size_t)blockIdx.x*256 + threadIdx.x;
  const int d  = (int)(i % Dn);
  const int tt = (int)((i/Dn) % Tn);
  x[i] += pe[tt*Dn+d] + lpe[tt*Dn+d];
}

// pack Wq/Wk/Wv [H,D,HD] -> P[D, 3*512] so QKV is one GEMM
__global__ void pack_qkv_kernel(const float* __restrict__ Wq, const float* __restrict__ Wk,
                                const float* __restrict__ Wv, float* __restrict__ P)
{
  const int i = blockIdx.x*256 + threadIdx.x;   // Dn*1536
  const int n = i % 1536, c = i / 1536;
  const float* W = (n<512) ? Wq : (n<1024 ? Wk : Wv);
  const int nn = n & 511, hh = nn>>5, d = nn&31;
  P[i] = W[((size_t)hh*Dn + c)*HDn + d];
}

// ---------------------------------------------------------------- attention
// one block per (b, h, 64-row query block); qkv f16 [B,T, 3*512] (q|k|v, head-major)
__global__ __launch_bounds__(256) void attn_kernel(const _Float16* __restrict__ qkv,
                                                   float* __restrict__ o)
{
  __shared__ float    S [64*392];   // raw scores then probs, 100352 B
  __shared__ _Float16 Qs[64*40];    // Q tile row-major
  __shared__ _Float16 Ks[384*40];   // K rows row-major (B-operand for Q K^T)
  __shared__ _Float16 Vt[32*392];   // V transposed [d][s] (B-operand for P V)
  const int blk = blockIdx.x;
  const int qb = blk % 6;
  const int h  = (blk/6) % Hn;
  const int b  = blk/(6*Hn);
  const int t = threadIdx.x, lane=t&31, w=t>>5;
  const _Float16* qbase = qkv + (size_t)b*Tn*1536 + h*HDn;
  const _Float16* kbase = qbase + 512;
  const _Float16* vbase = qbase + 1024;

  { // stage Q (64x32) -- straight f16 copy: async global->LDS DMA
    const int row=t>>2, seg=(t&3)*8;
    copy16_to_lds(&Qs[row*40+seg], qbase + (size_t)(qb*64+row)*1536 + seg);
  }
#pragma unroll
  for (int r=0;r<6;r++){ // stage K (384x32) -- async global->LDS DMA
    const int idx=t + r*256, row=idx>>2, seg=(idx&3)*8;
    copy16_to_lds(&Ks[row*40+seg], kbase + (size_t)row*1536 + seg);
  }
#pragma unroll
  for (int r=0;r<6;r++){ // stage V transposed (32x384): element-level, via VGPRs
    const int idx=t + r*256, s=idx>>2, dseg=(idx&3)*8;
    const v8h vv = *(const v8h*)(vbase + (size_t)s*1536 + dseg);
#pragma unroll
    for (int j=0;j<8;j++) Vt[(dseg+j)*392 + s] = vv[j];
  }
  async_lds_wait();     // drain ASYNCcnt before the barrier
  __syncthreads();

  { // S = Q K^T (raw; scale folded into softmax), causal tiles only
    const int wm = w & 3, js0 = w>>2;
    const v16h aq = load_a_frag(&Qs[(wm*16)*40], 40);
    const int lim = qb*64 + wm*16 + 15;
    const int hi=lane>>4, l=lane&15;
    for (int j=js0; j*16<=lim; j+=2){
      const v16h bk = load_b_frag(&Ks[(j*16)*40], 40);
      v8f s8 = {};
      s8 = wmma16(aq, bk, s8);
#pragma unroll
      for (int r=0;r<8;r++)
        S[(wm*16+hi*8+r)*392 + j*16 + l] = s8[r];
    }
  }
  __syncthreads();

  { // branchless causal softmax: 4 lanes/row, 96 cols each, vector LDS traffic
    const int row=t>>2, q4=t&3;
    const int cend = qb*64 + row + 1;     // valid columns [0, cend)
    const int c0 = q4*96;
    float* Sr = &S[row*392];
    const float scale = 0.04419417382415922f;  // 1/sqrt(512)
    float mx = -3.0e38f;
    for (int c=c0; c<c0+96; c+=4){
      const v4f v = *(const v4f*)&Sr[c];
#pragma unroll
      for (int j=0;j<4;j++){
        const float vv = v[j]*scale;
        mx = fmaxf(mx, (c+j<cend) ? vv : -3.0e38f);
      }
    }
    mx = fmaxf(mx, __shfl_xor(mx,1,32)); mx = fmaxf(mx, __shfl_xor(mx,2,32));
    float sum=0.f;
    for (int c=c0; c<c0+96; c+=4){
      const v4f v = *(const v4f*)&Sr[c];
      v4f p;
#pragma unroll
      for (int j=0;j<4;j++){
        const float e = __expf(v[j]*scale - mx);
        p[j] = (c+j<cend) ? e : 0.f;
        sum += p[j];
      }
      *(v4f*)&Sr[c] = p;
    }
    sum += __shfl_xor(sum,1,32); sum += __shfl_xor(sum,2,32);
    const float inv = 1.f/sum;
    for (int c=c0; c<c0+96; c+=4){
      v4f v = *(const v4f*)&Sr[c];
      v[0]*=inv; v[1]*=inv; v[2]*=inv; v[3]*=inv;
      *(v4f*)&Sr[c] = v;
    }
  }
  __syncthreads();

  { // O = P @ V  (wave -> one 16x16 output tile; K-chunks causally clipped)
    const int m2=w>>1, n2=w&1;
    v8f acc = {};
    const int kcmax = (qb*64 + m2*16 + 15) >> 5;
    for (int kc=0;kc<=kcmax;kc++){
      const v16h ap = load_a_frag_f32(&S[(m2*16)*392 + kc*32], 392);
      const v16h bv = load_b_frag(&Vt[(n2*16)*392 + kc*32], 392);
      acc = wmma16(ap, bv, acc);
    }
    const int hi=lane>>4, l=lane&15;
#pragma unroll
    for (int r=0;r<8;r++){
      const int trow = qb*64 + m2*16 + hi*8 + r;
      o[((size_t)b*Tn + trow)*Dn + h*HDn + n2*16 + l] = acc[r];
    }
  }
}

// ---------------------------------------------------------------- MSE loss (deterministic)
__global__ __launch_bounds__(256) void loss_partial_kernel(
    const float* __restrict__ logits, const float* __restrict__ tgt,
    float* __restrict__ part)
{
  __shared__ float red[8];
  const int t=threadIdx.x, lane=t&31, w=t>>5;
  float s=0.f;
#pragma unroll
  for (int r=0;r<6;r++){
    const size_t idx=(size_t)blockIdx.x*1536 + r*256 + t;
    const float d = logits[idx]-tgt[idx];
    s += d*d;
  }
#pragma unroll
  for (int o=16;o;o>>=1) s += __shfl_xor(s,o,32);
  if (lane==0) red[w]=s;
  __syncthreads();
  if (t==0){
    float tot=0.f;
#pragma unroll
    for (int i=0;i<8;i++) tot+=red[i];
    part[blockIdx.x]=tot;
  }
}

__global__ __launch_bounds__(256) void loss_final_kernel(
    const float* __restrict__ part, float* __restrict__ out)
{
  __shared__ float red[8];
  const int t=threadIdx.x, lane=t&31, w=t>>5;
  float s=0.f;
#pragma unroll
  for (int r=0;r<4;r++) s += part[t + r*256];
#pragma unroll
  for (int o=16;o;o>>=1) s += __shfl_xor(s,o,32);
  if (lane==0) red[w]=s;
  __syncthreads();
  if (t==0){
    float tot=0.f;
#pragma unroll
    for (int i=0;i<8;i++) tot+=red[i];
    out[0] = tot * (1.0f/((float)Mrows*(float)CFn));
  }
}

// ---------------------------------------------------------------- host
extern "C" void kernel_launch(void* const* d_in, const int* in_sizes, int n_in,
                              void* d_out, int out_size, void* d_ws, size_t ws_size,
                              hipStream_t stream)
{
  (void)in_sizes; (void)n_in; (void)out_size; (void)ws_size;
  const float* index   = (const float*)d_in[0];
  const float* targets = (const float*)d_in[1];
  const float* W_embed = (const float*)d_in[2];
  const float* b_embed = (const float*)d_in[3];
  const float* pe      = (const float*)d_in[4];
  const float* lpe     = (const float*)d_in[5];
  const float* Wq      = (const float*)d_in[6];
  const float* Wk      = (const float*)d_in[7];
  const float* Wv      = (const float*)d_in[8];
  const float* Wproj   = (const float*)d_in[9];
  const float* bproj   = (const float*)d_in[10];
  const float* ln1g    = (const float*)d_in[11];
  const float* ln1b    = (const float*)d_in[12];
  const float* ln2g    = (const float*)d_in[13];
  const float* ln2b    = (const float*)d_in[14];
  const float* W1      = (const float*)d_in[15];
  const float* b1      = (const float*)d_in[16];
  const float* W2      = (const float*)d_in[17];
  const float* b2      = (const float*)d_in[18];
  const float* Wfin    = (const float*)d_in[19];
  const float* bfin    = (const float*)d_in[20];

  char* ws = (char*)d_ws;
  size_t off = 0;
  auto carve = [&](size_t bytes)->char*{
    char* p = ws + off; off = (off + bytes + 255) & ~size_t(255); return p;
  };
  float*    x    = (float*)   carve((size_t)Mrows*Dn*4);     // activations
  float*    hbuf = (float*)   carve((size_t)Mrows*Dn*4);     // LN output
  _Float16* qkv  = (_Float16*)carve((size_t)Mrows*1536*2);   // q|k|v f16
  float*    ao   = (float*)   carve((size_t)Mrows*Dn*4);     // attention out
  float*    ffn  = (float*)   carve((size_t)Mrows*HIDn*4);   // FFN hidden
  float*    Wpk  = (float*)   carve((size_t)Dn*1536*4);      // packed QKV weights
  float*    part = (float*)   carve(1024*4);                 // loss partials

  float* logits = (float*)d_out;
  float* loss   = logits + (size_t)Mrows*CFn;

  // embed + positional encodings
  gemm_wmma<float,false,false><<<dim3(Dn/128, Mrows/128), 256, 0, stream>>>(
      index, W_embed, b_embed, nullptr, x, Mrows, Dn, CFn);
  addpe_kernel<<<(Mrows*Dn)/256, 256, 0, stream>>>(x, pe, lpe);

  for (int l=0;l<Ln;l++){
    const size_t wo = (size_t)l*Hn*Dn*HDn;            // 262144 per layer
    ln_kernel<<<Mrows/8,256,0,stream>>>(x, ln1g+l*Dn, ln1b+l*Dn, hbuf);
    pack_qkv_kernel<<<(Dn*1536)/256,256,0,stream>>>(Wq+wo, Wk+wo, Wv+wo, Wpk);
    gemm_wmma<_Float16,false,false><<<dim3(1536/128, Mrows/128),256,0,stream>>>(
        hbuf, Wpk, nullptr, nullptr, qkv, Mrows, 1536, Dn);
    attn_kernel<<<Bn*Hn*(Tn/64),256,0,stream>>>(qkv, ao);
    gemm_wmma<float,false,true><<<dim3(Dn/128,Mrows/128),256,0,stream>>>(
        ao, Wproj+(size_t)l*Dn*Dn, bproj+l*Dn, x, x, Mrows, Dn, Dn);
    ln_kernel<<<Mrows/8,256,0,stream>>>(x, ln2g+l*Dn, ln2b+l*Dn, hbuf);
    gemm_wmma<float,true,false><<<dim3(HIDn/128,Mrows/128),256,0,stream>>>(
        hbuf, W1+(size_t)l*Dn*HIDn, b1+l*HIDn, nullptr, ffn, Mrows, HIDn, Dn);
    gemm_wmma<float,false,true><<<dim3(Dn/128,Mrows/128),256,0,stream>>>(
        ffn, W2+(size_t)l*HIDn*Dn, b2+l*Dn, x, x, Mrows, Dn, HIDn);
  }

  gemm_wmma<float,false,false><<<dim3(CFn/128,Mrows/128),256,0,stream>>>(
      x, Wfin, bfin, nullptr, logits, Mrows, CFn, Dn);
  loss_partial_kernel<<<1024,256,0,stream>>>(logits, targets, part);
  loss_final_kernel<<<1,256,0,stream>>>(part, loss);
}
